// MultiHeadAttention_48833778155621
// MI455X (gfx1250) — matrix operand order repaired and checked by reading, not compiled, not run on hardware
//
#include <hip/hip_runtime.h>
#include <stdint.h>

// ---------------------------------------------------------------------------
// MHA forward for MI455X (gfx1250, wave32, WMMA).
// Stage 0: fp32 -> bf16 pre-conversion (activations+weights) + mask->float bias.
// Stage 1: 3x projection GEMM (pure bf16 WMMA, 32x64 tile/wave), Q pre-scaled
//          by 1/sqrt(dk), Q/K head-split, V transposed.
// Stage 2: fused flash attention, 64-key chunks (16 WMMA/chunk), ds_swizzle
//          xor-reductions, K/V L2-resident read directly from global; LDS only
//          for the P C-layout -> A-layout shuffle.
// Stage 3: output projection GEMM (32x64 tile/wave) -> fp32.
// ---------------------------------------------------------------------------

#define D_MODEL   1024
#define NUM_HEADS 16
#define D_K       64
#define BB        2
#define SS        2048
#define MROWS     (BB * SS)          // 4096 rows for the big GEMMs

typedef __attribute__((ext_vector_type(16))) __bf16    v16bf;
typedef __attribute__((ext_vector_type(8)))  float     v8f;
typedef __attribute__((ext_vector_type(4)))  unsigned  u32x4;

struct frag_bits { u32x4 lo, hi; };   // 32 bytes == one v16bf fragment

// ---------------- bf16 helpers (round-to-nearest-even) ----------------------
__device__ inline unsigned short f2bfu(float f) {
  union { float f; unsigned u; } x; x.f = f;
  unsigned r = x.u + 0x7fffu + ((x.u >> 16) & 1u);
  return (unsigned short)(r >> 16);
}

// ---------------- cross-lane xor exchange via ds_swizzle ---------------------
// Group-of-32 swizzle: lane' = ((lane & and) | or) ^ xor; and=0x1f, or=0.
template<int XOR>
__device__ inline float lane_xor(float v) {
  int r = __builtin_amdgcn_ds_swizzle(__builtin_bit_cast(int, v), (XOR << 10) | 0x1f);
  return __builtin_bit_cast(float, r);
}
// Reduce within each 16-lane half (rows live per-half in the C layout).
__device__ inline float red_max16(float t) {
  t = fmaxf(t, lane_xor<1>(t));
  t = fmaxf(t, lane_xor<2>(t));
  t = fmaxf(t, lane_xor<4>(t));
  t = fmaxf(t, lane_xor<8>(t));
  return t;
}
__device__ inline float red_add16(float t) {
  t += lane_xor<1>(t);
  t += lane_xor<2>(t);
  t += lane_xor<4>(t);
  t += lane_xor<8>(t);
  return t;
}

// ---------------- fragment loaders (pure b128 + bit_cast, no per-elem VALU) --
// A-matrix 16x32 bf16: lane holds row M=lane%16, half=lane/16.
// Element e has K = e + 16*half (16 contiguous, matching loadB_bf)
// -> exactly two 16B loads.
__device__ inline v16bf loadA_bf(const unsigned short* src, int half) {
  frag_bits f;
  f.lo = *(const u32x4*)(src + 16 * half);
  f.hi = *(const u32x4*)(src + 16 * half + 8);
  return __builtin_bit_cast(v16bf, f);
}
// B-matrix 32x16 bf16: lane holds col N=lane%16, half=lane/16; element e has
// K = e + 16*half. src must already include the +16*half offset: 16 contiguous.
__device__ inline v16bf loadB_bf(const unsigned short* src) {
  frag_bits f;
  f.lo = *(const u32x4*)(src);
  f.hi = *(const u32x4*)(src + 8);
  return __builtin_bit_cast(v16bf, f);
}

#define WMMA_BF16(A, B, C) \
  __builtin_amdgcn_wmma_f32_16x16x32_bf16(false, (A), false, (B), (short)0, (C), false, false)

// ---------------------------------------------------------------------------
// Stage 0a: streaming fp32 -> bf16 conversion, 4 elements per thread.
// ---------------------------------------------------------------------------
__global__ __launch_bounds__(256)
void cvt_f32_to_bf16(const float* __restrict__ src, unsigned short* __restrict__ dst,
                     int n4) {
  const int i = blockIdx.x * 256 + threadIdx.x;
  if (i < n4) {
    const float4 f = ((const float4*)src)[i];
    uint2 o;
    o.x = (unsigned)f2bfu(f.x) | ((unsigned)f2bfu(f.y) << 16);
    o.y = (unsigned)f2bfu(f.z) | ((unsigned)f2bfu(f.w) << 16);
    ((uint2*)dst)[i] = o;
  }
}

// Stage 0b: int mask -> additive float bias (0 or -inf).
__global__ __launch_bounds__(256)
void cvt_mask_bias(const int* __restrict__ mask, float* __restrict__ mbias, int n) {
  const int i = blockIdx.x * 256 + threadIdx.x;
  if (i < n) mbias[i] = mask[i] ? 0.f : -__builtin_inff();
}

// ---------------------------------------------------------------------------
// Stage 1/3: Y = (X * W^T + b) * oscale with bf16 A/B, fp32 accumulate.
// 32x64 output tile per wave: 2 A-frags x 4 B-frags -> 8 WMMAs per 12 b128
// loads per 32-wide K step (good VMEM:WMMA issue ratio).
// MODE 0: bf16 out, head-split (B,H,S,64)             (Q, K projections)
// MODE 1: bf16 out, head-split transposed (B,H,64,S)  (V projection)
// MODE 2: fp32 out, row-major (M,1024) + bias         (output projection)
// ---------------------------------------------------------------------------
template<int MODE>
__global__ __launch_bounds__(128)
void gemm_bf16_kernel(const unsigned short* __restrict__ Xbf,
                      const unsigned short* __restrict__ Wbf,
                      const float* __restrict__ bias,
                      void* __restrict__ outp, float oscale) {
  const int lane = threadIdx.x & 31;
  const int wave = threadIdx.x >> 5;
  const int tile = blockIdx.x * 4 + wave;        // (4096/32)*(1024/64) = 2048 tiles
  const int tm = tile / (D_MODEL / 64);          // 0..127
  const int tn = tile % (D_MODEL / 64);          // 0..15
  const int half = lane >> 4;
  const int l16  = lane & 15;

  const unsigned short* xrow0 = Xbf + (size_t)(tm * 32 + l16) * D_MODEL;
  const unsigned short* xrow1 = xrow0 + 16 * D_MODEL;
  const unsigned short* wrow0 = Wbf + (size_t)(tn * 64 + 0 * 16 + l16) * D_MODEL + 16 * half;
  const unsigned short* wrow1 = Wbf + (size_t)(tn * 64 + 1 * 16 + l16) * D_MODEL + 16 * half;
  const unsigned short* wrow2 = Wbf + (size_t)(tn * 64 + 2 * 16 + l16) * D_MODEL + 16 * half;
  const unsigned short* wrow3 = Wbf + (size_t)(tn * 64 + 3 * 16 + l16) * D_MODEL + 16 * half;

  v8f acc[2][4];
#pragma unroll
  for (int i = 0; i < 2; ++i)
#pragma unroll
    for (int t = 0; t < 4; ++t) acc[i][t] = (v8f){};

  for (int k0 = 0; k0 < D_MODEL; k0 += 32) {
    const v16bf a0 = loadA_bf(xrow0 + k0, half);
    const v16bf a1 = loadA_bf(xrow1 + k0, half);
    const v16bf b0 = loadB_bf(wrow0 + k0);
    const v16bf b1 = loadB_bf(wrow1 + k0);
    const v16bf b2 = loadB_bf(wrow2 + k0);
    const v16bf b3 = loadB_bf(wrow3 + k0);
    acc[0][0] = WMMA_BF16(a0, b0, acc[0][0]);
    acc[1][0] = WMMA_BF16(a1, b0, acc[1][0]);
    acc[0][1] = WMMA_BF16(a0, b1, acc[0][1]);
    acc[1][1] = WMMA_BF16(a1, b1, acc[1][1]);
    acc[0][2] = WMMA_BF16(a0, b2, acc[0][2]);
    acc[1][2] = WMMA_BF16(a1, b2, acc[1][2]);
    acc[0][3] = WMMA_BF16(a0, b3, acc[0][3]);
    acc[1][3] = WMMA_BF16(a1, b3, acc[1][3]);
  }

#pragma unroll
  for (int t = 0; t < 4; ++t) {
    const int n = tn * 64 + t * 16 + l16;        // C layout: col = lane%16
    const float bn = bias[n];
    const int h = n / D_K, d = n % D_K;
#pragma unroll
    for (int i = 0; i < 2; ++i) {
#pragma unroll
      for (int j = 0; j < 8; ++j) {
        const int row = tm * 32 + i * 16 + j + 8 * half;  // C layout: M = slot+8*half
        const float v = (acc[i][t][j] + bn) * oscale;
        if (MODE == 2) {
          ((float*)outp)[(size_t)row * D_MODEL + n] = v;
        } else {
          const int b = row / SS, s = row % SS;
          const size_t idx = (MODE == 0)
              ? (((size_t)(b * NUM_HEADS + h)) * SS + s) * D_K + d     // (B,H,S,dk)
              : (((size_t)(b * NUM_HEADS + h)) * D_K + d) * SS + s;    // (B,H,dk,S)
          ((unsigned short*)outp)[idx] = f2bfu(v);
        }
      }
    }
  }
}

// ---------------------------------------------------------------------------
// Stage 2: fused flash attention. One wave owns 16 query rows of one (b,h),
// iterating over S in 64-key chunks (4 score tiles + 8 PV WMMAs per chunk,
// so the serialized per-row xor-reductions amortize over 2x more keys).
// K (B,H,S,64) and Vt (B,H,64,S) are bf16 and L2-resident (16MB vs 192MB L2)
// -> fragments load directly from global as contiguous b128 pairs. Q is
// pre-scaled by 1/sqrt(dk); mask is an additive float bias. LDS is used only
// for the P C-layout -> A-layout shuffle (wave-private, DS in-order).
// ---------------------------------------------------------------------------
__global__ __launch_bounds__(128)
void flash_attn_kernel(const unsigned short* __restrict__ Qbf,
                       const unsigned short* __restrict__ Kbf,
                       const unsigned short* __restrict__ Vt,
                       const float* __restrict__ mbias,
                       unsigned short* __restrict__ Cbf) {
  __shared__ __align__(16) unsigned short plds[4][16 * 64];   // 2KB per wave

  const int lane = threadIdx.x & 31;
  const int wave = threadIdx.x >> 5;
  const int w  = blockIdx.x * 4 + wave;          // 4096 wave-tiles
  const int qt = w % (SS / 16);
  const int bh = w / (SS / 16);
  const int b  = bh / NUM_HEADS;
  const int h  = bh % NUM_HEADS;
  const int half = lane >> 4;
  const int l16  = lane & 15;
  const float NEG_INF = -__builtin_inff();

  // Q fragments for this q-tile (two K-halves of d_k = 64), pre-loaded once.
  const unsigned short* qrow = Qbf + ((size_t)bh * SS + qt * 16 + l16) * D_K;
  const v16bf qa0 = loadA_bf(qrow, half);
  const v16bf qa1 = loadA_bf(qrow + 32, half);

  float mrow[8], lrow[8];
#pragma unroll
  for (int j = 0; j < 8; ++j) { mrow[j] = NEG_INF; lrow[j] = 0.f; }
  v8f acc[4];
#pragma unroll
  for (int t = 0; t < 4; ++t) acc[t] = (v8f){};

  unsigned short* pl = &plds[wave][0];
  const unsigned short* krow0 = Kbf + ((size_t)bh * SS + l16) * D_K + 16 * half;
  const unsigned short* vrow0 = Vt + ((size_t)bh * D_K + l16) * SS + 16 * half;
  const float* mrow0 = mbias + b * SS;

  for (int kt = 0; kt < SS; kt += 64) {
    // prefetch next chunk's K/V streams (global_prefetch_b8)
    __builtin_prefetch((const void*)(krow0 + (size_t)(kt + 64) * D_K), 0, 3);
    __builtin_prefetch((const void*)(vrow0 + kt + 64), 0, 3);

    // ---- scores: S = Q(16x64) . K^T(64x64) -> four 16x16 C tiles
    v8f c[4];
#pragma unroll
    for (int tt = 0; tt < 4; ++tt) {
      const unsigned short* kp = krow0 + (size_t)(kt + tt * 16) * D_K;
      v8f cc = {};
      cc = WMMA_BF16(qa0, loadB_bf(kp), cc);         // d 0..31 slice
      cc = WMMA_BF16(qa1, loadB_bf(kp + 32), cc);    // d 32..63 slice
      c[tt] = cc;
    }

    // ---- additive mask bias (0 or -inf), one add per score element
    float s[4][8];
#pragma unroll
    for (int tt = 0; tt < 4; ++tt) {
      const float mb = mrow0[kt + tt * 16 + l16];
#pragma unroll
      for (int j = 0; j < 8; ++j) s[tt][j] = c[tt][j] + mb;
    }

    // ---- online softmax update, per row (row = slot j + 8*half)
#pragma unroll
    for (int j = 0; j < 8; ++j) {
      float t = fmaxf(fmaxf(s[0][j], s[1][j]), fmaxf(s[2][j], s[3][j]));
      t = red_max16(t);
      const float mnew = fmaxf(mrow[j], t);
      const bool dead = (mnew == NEG_INF);       // row fully masked so far
      const float sc = dead ? 1.f : __expf(mrow[j] - mnew);
      const float p0 = dead ? 0.f : __expf(s[0][j] - mnew);
      const float p1 = dead ? 0.f : __expf(s[1][j] - mnew);
      const float p2 = dead ? 0.f : __expf(s[2][j] - mnew);
      const float p3 = dead ? 0.f : __expf(s[3][j] - mnew);
      const float ps = red_add16((p0 + p1) + (p2 + p3));
      lrow[j] = lrow[j] * sc + ps;
      mrow[j] = mnew;
      acc[0][j] *= sc; acc[1][j] *= sc; acc[2][j] *= sc; acc[3][j] *= sc;
      // stash P in LDS (C layout -> row-major 16x64) for A-fragment reload
      unsigned short* pr = pl + (j + 8 * half) * 64 + l16;
      pr[0]  = f2bfu(p0);
      pr[16] = f2bfu(p1);
      pr[32] = f2bfu(p2);
      pr[48] = f2bfu(p3);
    }

    // ---- P (16x64) as two A fragments; Vt gives contiguous B fragments
    const v16bf pa0 = loadA_bf(pl + l16 * 64, half);       // keys kt..kt+31
    const v16bf pa1 = loadA_bf(pl + l16 * 64 + 32, half);  // keys kt+32..kt+63
#pragma unroll
    for (int t4 = 0; t4 < 4; ++t4) {
      const unsigned short* vp = vrow0 + (size_t)t4 * 16 * SS + kt;  // d tile rows
      acc[t4] = WMMA_BF16(pa0, loadB_bf(vp), acc[t4]);         // keys kt..+31
      acc[t4] = WMMA_BF16(pa1, loadB_bf(vp + 32), acc[t4]);    // keys kt+32..+63
    }
  }

  // ---- normalize and write context in concat layout (b, s, h*64+d), bf16
#pragma unroll
  for (int j = 0; j < 8; ++j) {
    const float inv = (lrow[j] > 0.f) ? (1.f / lrow[j]) : 0.f;  // nan_to_num path
    const int ss = qt * 16 + j + 8 * half;
    const size_t base = ((size_t)b * SS + ss) * D_MODEL + h * D_K;
    Cbf[base + 0 * 16 + l16] = f2bfu(acc[0][j] * inv);
    Cbf[base + 1 * 16 + l16] = f2bfu(acc[1][j] * inv);
    Cbf[base + 2 * 16 + l16] = f2bfu(acc[2][j] * inv);
    Cbf[base + 3 * 16 + l16] = f2bfu(acc[3][j] * inv);
  }
}

// ---------------------------------------------------------------------------
extern "C" void kernel_launch(void* const* d_in, const int* in_sizes, int n_in,
                              void* d_out, int out_size, void* d_ws, size_t ws_size,
                              hipStream_t stream) {
  (void)in_sizes; (void)n_in; (void)out_size; (void)ws_size;
  const float* query = (const float*)d_in[0];
  const float* key_  = (const float*)d_in[1];
  const float* value = (const float*)d_in[2];
  const int*   mask  = (const int*)d_in[3];
  const float* Wq = (const float*)d_in[4];
  const float* bq = (const float*)d_in[5];
  const float* Wk = (const float*)d_in[6];
  const float* bk = (const float*)d_in[7];
  const float* Wv = (const float*)d_in[8];
  const float* bv = (const float*)d_in[9];
  const float* Wo = (const float*)d_in[10];
  const float* bo = (const float*)d_in[11];

  // Workspace layout (ushorts): 3 bf16 inputs (4M each) + 4 bf16 weights
  // (1M each) + Q/K/Vt/C (4M each) = 32M ushorts = 64MB, + 16KB mask bias.
  const size_t NX = (size_t)MROWS * D_MODEL;     // 4,194,304
  const size_t NW = (size_t)D_MODEL * D_MODEL;   // 1,048,576
  unsigned short* Xq  = (unsigned short*)d_ws;
  unsigned short* Xk  = Xq  + NX;
  unsigned short* Xv  = Xk  + NX;
  unsigned short* Wqb = Xv  + NX;
  unsigned short* Wkb = Wqb + NW;
  unsigned short* Wvb = Wkb + NW;
  unsigned short* Wob = Wvb + NW;
  unsigned short* Qbf = Wob + NW;
  unsigned short* Kbf = Qbf + NX;
  unsigned short* Vtb = Kbf + NX;
  unsigned short* Cbf = Vtb + NX;
  float*          Mbf = (float*)(Cbf + NX);      // BB*SS floats

  const dim3 blk(128);
  const dim3 cblk(256);

  // Stage 0: bf16 pre-conversion + mask bias (memory-bound streaming).
  cvt_f32_to_bf16<<<(int)(NX / 4 / 256), cblk, 0, stream>>>(query, Xq, (int)(NX / 4));
  cvt_f32_to_bf16<<<(int)(NX / 4 / 256), cblk, 0, stream>>>(key_,  Xk, (int)(NX / 4));
  cvt_f32_to_bf16<<<(int)(NX / 4 / 256), cblk, 0, stream>>>(value, Xv, (int)(NX / 4));
  cvt_f32_to_bf16<<<(int)(NW / 4 / 256), cblk, 0, stream>>>(Wq, Wqb, (int)(NW / 4));
  cvt_f32_to_bf16<<<(int)(NW / 4 / 256), cblk, 0, stream>>>(Wk, Wkb, (int)(NW / 4));
  cvt_f32_to_bf16<<<(int)(NW / 4 / 256), cblk, 0, stream>>>(Wv, Wvb, (int)(NW / 4));
  cvt_f32_to_bf16<<<(int)(NW / 4 / 256), cblk, 0, stream>>>(Wo, Wob, (int)(NW / 4));
  cvt_mask_bias<<<(BB * SS) / 256, cblk, 0, stream>>>(mask, Mbf, BB * SS);

  // Stage 1: projections. (4096/32)*(1024/64) = 2048 wave-tiles, 4/block.
  // Q pre-scaled by 1/sqrt(d_k) = 0.125 so attention scores need no scaling.
  gemm_bf16_kernel<0><<<512, blk, 0, stream>>>(Xq, Wqb, bq, Qbf, 0.125f);
  gemm_bf16_kernel<0><<<512, blk, 0, stream>>>(Xk, Wkb, bk, Kbf, 1.0f);
  gemm_bf16_kernel<1><<<512, blk, 0, stream>>>(Xv, Wvb, bv, Vtb, 1.0f);

  // Stage 2: fused attention. 2*16*128 = 4096 q-tiles, 4 waves/block.
  flash_attn_kernel<<<1024, blk, 0, stream>>>(Qbf, Kbf, Vtb, Mbf, Cbf);

  // Stage 3: output projection -> fp32.
  gemm_bf16_kernel<2><<<512, blk, 0, stream>>>(Cbf, Wob, bo, (float*)d_out, 1.0f);
}
